// TorchLMHeadPreference_67920612819620
// MI455X (gfx1250) — matrix operand order repaired and checked by reading, not compiled, not run on hardware
//
#include <hip/hip_runtime.h>
#include <hip/hip_bf16.h>
#include <math.h>

typedef __attribute__((ext_vector_type(16))) __bf16 v16bf;
typedef __attribute__((ext_vector_type(8)))  float  v8f;
typedef __attribute__((ext_vector_type(4)))  float  f32x4;

#define H_DIM   1024
#define ROWS    2048          // B*T = 4*512
#define T_LEN   512
#define B_SEQ   4
#define VOCAB   128000
#define BM      64            // rows per workgroup
#define BN      256           // vocab cols per workgroup
#define KSTEP   32
#define NVB     (VOCAB / BN)  // 500 vocab blocks
#define MBLK    (ROWS / BM)   // 32 row blocks
#define CAP     30.0f
#define IGN     (-100)
#define BETA_C  0.1f

// ---------------------------------------------------------------------------
// Kernel 1: bf16 WMMA GEMM tile + softcap + target-logit capture + per-row
// online-softmax partials (max, sumexp) for this vocab slice.
// Workgroup: 256 threads = 8 waves in a 2x4 grid; wave tile = 32(M) x 64(N).
// ---------------------------------------------------------------------------
__global__ __launch_bounds__(256)
void lmhead_gemm_partials(const float* __restrict__ x,
                          const float* __restrict__ W,
                          const int*   __restrict__ target,
                          float* __restrict__ part_m,
                          float* __restrict__ part_s,
                          float* __restrict__ t_logit)
{
    const int tid   = threadIdx.x;
    const int lane  = tid & 31;
    const int wid   = tid >> 5;       // 0..7
    const int waveM = wid >> 2;       // 0..1
    const int waveN = wid & 3;        // 0..3
    const int mblk  = blockIdx.x;     // 0..31  (fast dim -> L2 reuse of W tile)
    const int vblk  = blockIdx.y;     // 0..499
    const int row0  = mblk * BM;
    const int col0  = vblk * BN;

    __shared__ int   s_label[BM];
    __shared__ float s_stat[4][BM][2];   // [waveN][local row][{max,sumexp}]

    if (tid < BM) {
        int t = target[row0 + tid];
        s_label[tid] = (t == IGN) ? 0 : t;
    }
    __syncthreads();

    const int half = lane >> 4;    // 0/1: K-half selector per ISA bf16 layouts
    const int l15  = lane & 15;

    const int rbase = row0 + waveM * 32;   // global row of wave tile
    const int cbase = col0 + waveN * 64;   // global col of wave tile

    v8f acc[2][4];
    const v8f vzero = {0.f,0.f,0.f,0.f,0.f,0.f,0.f,0.f};
    #pragma unroll
    for (int i = 0; i < 2; i++)
        #pragma unroll
        for (int j = 0; j < 4; j++) acc[i][j] = vzero;

    for (int k0 = 0; k0 < H_DIM; k0 += KSTEP) {
        // ---- A fragments: 16x32 bf16 (ISA layout: lanes 0-15 K0-7/K16-23,
        //      lanes 16-31 K8-15/K24-31), from fp32 x, converted in-register.
        v16bf afrag[2];
        #pragma unroll
        for (int mi = 0; mi < 2; mi++) {
            const float* pa = x + (size_t)(rbase + mi*16 + l15) * H_DIM + k0 + half*8;
            const f32x4* pa4 = (const f32x4*)pa;
            f32x4 f0 = pa4[0], f1 = pa4[1], f2 = pa4[4], f3 = pa4[5];
            v16bf a;
            #pragma unroll
            for (int e = 0; e < 4; e++) {
                a[e]      = (__bf16)f0[e];
                a[4 + e]  = (__bf16)f1[e];
                a[8 + e]  = (__bf16)f2[e];
                a[12 + e] = (__bf16)f3[e];
            }
            afrag[mi] = a;
        }
        // ---- B fragments: 32x16 bf16 (lane = column = W row; same K layout as A)
        v16bf bfrag[4];
        #pragma unroll
        for (int ni = 0; ni < 4; ni++) {
            const float* pb = W + (size_t)(cbase + ni*16 + l15) * H_DIM + k0 + half*8;
            const f32x4* pb4 = (const f32x4*)pb;
            f32x4 g0 = pb4[0], g1 = pb4[1], g2 = pb4[4], g3 = pb4[5];
            v16bf b;
            #pragma unroll
            for (int e = 0; e < 4; e++) {
                b[e]      = (__bf16)g0[e];
                b[4 + e]  = (__bf16)g1[e];
                b[8 + e]  = (__bf16)g2[e];
                b[12 + e] = (__bf16)g3[e];
            }
            bfrag[ni] = b;
            // branchless speculative prefetch of next K tile of W
            // (global_prefetch_b8; invalid-address prefetches are dropped)
            __builtin_prefetch(pb + KSTEP, 0, 3);
        }
        // ---- 8 WMMAs, f32 accumulate
        #pragma unroll
        for (int mi = 0; mi < 2; mi++)
            #pragma unroll
            for (int ni = 0; ni < 4; ni++)
                acc[mi][ni] = __builtin_amdgcn_wmma_f32_16x16x32_bf16(
                    false, afrag[mi], false, bfrag[ni],
                    (short)0, acc[mi][ni], false, false);
    }

    // ---- Epilogue: softcap, target capture, row max / sumexp for this slice
    const float invcap = 1.0f / CAP;
    #pragma unroll
    for (int mi = 0; mi < 2; mi++) {
        #pragma unroll
        for (int ni = 0; ni < 4; ni++) {
            #pragma unroll
            for (int g = 0; g < 8; g++) {
                float l = acc[mi][ni][g];
                l = CAP * tanhf(l * invcap);
                acc[mi][ni][g] = l;
                // C/D layout: VGPR g = row (g + 8*half), lane&15 = col
                int rl = waveM*32 + mi*16 + g + 8*half;   // local row 0..63
                int cg = cbase + ni*16 + l15;             // global vocab col
                if (s_label[rl] == cg) t_logit[row0 + rl] = l;
            }
        }
        #pragma unroll
        for (int g = 0; g < 8; g++) {
            float m = fmaxf(fmaxf(acc[mi][0][g], acc[mi][1][g]),
                            fmaxf(acc[mi][2][g], acc[mi][3][g]));
            #pragma unroll
            for (int s = 1; s < 16; s <<= 1)
                m = fmaxf(m, __shfl_xor(m, s, 32));   // 16 lanes share a row
            float e = 0.0f;
            #pragma unroll
            for (int ni = 0; ni < 4; ni++)
                e += expf(acc[mi][ni][g] - m);
            #pragma unroll
            for (int s = 1; s < 16; s <<= 1)
                e += __shfl_xor(e, s, 32);
            if (l15 == 0) {
                int rl = waveM*32 + mi*16 + g + 8*half;
                s_stat[waveN][rl][0] = m;
                s_stat[waveN][rl][1] = e;
            }
        }
    }
    __syncthreads();

    if (tid < BM) {
        float m0 = s_stat[0][tid][0], m1 = s_stat[1][tid][0];
        float m2 = s_stat[2][tid][0], m3 = s_stat[3][tid][0];
        float M = fmaxf(fmaxf(m0, m1), fmaxf(m2, m3));
        float S = s_stat[0][tid][1] * expf(m0 - M)
                + s_stat[1][tid][1] * expf(m1 - M)
                + s_stat[2][tid][1] * expf(m2 - M)
                + s_stat[3][tid][1] * expf(m3 - M);
        size_t idx = (size_t)(row0 + tid) * NVB + vblk;
        part_m[idx] = M;
        part_s[idx] = S;
    }
}

// ---------------------------------------------------------------------------
// Kernel 2: one wave per token row; online-merge 500 (m,s) partials,
// emit masked per-token logp.
// ---------------------------------------------------------------------------
__global__ __launch_bounds__(256)
void lmhead_reduce_rows(const float* __restrict__ part_m,
                        const float* __restrict__ part_s,
                        const float* __restrict__ t_logit,
                        const int*   __restrict__ target,
                        float* __restrict__ wlogp,
                        float* __restrict__ wmask)
{
    const int row  = blockIdx.x * 8 + (threadIdx.x >> 5);
    const int lane = threadIdx.x & 31;
    if (row >= ROWS) return;

    float m = -1e30f, s = 0.0f;
    for (int i = lane; i < NVB; i += 32) {
        float pm = part_m[(size_t)row * NVB + i];
        float ps = part_s[(size_t)row * NVB + i];
        float nm = fmaxf(m, pm);
        s = s * expf(m - nm) + ps * expf(pm - nm);
        m = nm;
    }
    #pragma unroll
    for (int x = 1; x < 32; x <<= 1) {
        float om = __shfl_xor(m, x, 32);
        float os = __shfl_xor(s, x, 32);
        float nm = fmaxf(m, om);
        s = s * expf(m - nm) + os * expf(om - nm);
        m = nm;
    }
    if (lane == 0) {
        float lp = t_logit[row] - m - logf(s);
        bool  ok = (target[row] != IGN);
        wlogp[row] = ok ? lp : 0.0f;
        wmask[row] = ok ? 1.0f : 0.0f;
    }
}

// ---------------------------------------------------------------------------
// Kernel 3: per-sequence masked average of logp over T.
// ---------------------------------------------------------------------------
__global__ __launch_bounds__(256)
void lmhead_seq_avg(const float* __restrict__ wlogp,
                    const float* __restrict__ wmask,
                    float* __restrict__ seqavg)
{
    __shared__ float ssum[256];
    __shared__ float scnt[256];
    const int b = blockIdx.x;
    float s = 0.0f, c = 0.0f;
    for (int t = threadIdx.x; t < T_LEN; t += 256) {
        s += wlogp[b * T_LEN + t];
        c += wmask[b * T_LEN + t];
    }
    ssum[threadIdx.x] = s;
    scnt[threadIdx.x] = c;
    __syncthreads();
    for (int o = 128; o > 0; o >>= 1) {
        if (threadIdx.x < o) {
            ssum[threadIdx.x] += ssum[threadIdx.x + o];
            scnt[threadIdx.x] += scnt[threadIdx.x + o];
        }
        __syncthreads();
    }
    if (threadIdx.x == 0) seqavg[b] = ssum[0] / scnt[0];
}

// ---------------------------------------------------------------------------
// Kernel 4: final scalar preference loss. len_chosen = B/2 = 2.
// ---------------------------------------------------------------------------
__global__ void lmhead_final(const float* __restrict__ seqavg,
                             float* __restrict__ out)
{
    if (threadIdx.x == 0 && blockIdx.x == 0) {
        float d = (seqavg[0] - seqavg[2]) + (seqavg[1] - seqavg[3]);
        out[0] = -BETA_C * 0.5f * d;
    }
}

// ---------------------------------------------------------------------------
extern "C" void kernel_launch(void* const* d_in, const int* in_sizes, int n_in,
                              void* d_out, int out_size, void* d_ws, size_t ws_size,
                              hipStream_t stream)
{
    const float* x      = (const float*)d_in[0];   // (4,512,1024) f32
    const float* W      = (const float*)d_in[1];   // (128000,1024) f32
    const int*   target = (const int*)d_in[2];     // (4,512) int

    float* part_m  = (float*)d_ws;                     // ROWS*NVB
    float* part_s  = part_m + (size_t)ROWS * NVB;      // ROWS*NVB
    float* t_logit = part_s + (size_t)ROWS * NVB;      // ROWS
    float* wlogp   = t_logit + ROWS;                   // ROWS
    float* wmask   = wlogp + ROWS;                     // ROWS
    float* seqavg  = wmask + ROWS;                     // B_SEQ

    dim3 grid1(MBLK, NVB);     // x fast -> W v-tile shared in L2 across M blocks
    lmhead_gemm_partials<<<grid1, 256, 0, stream>>>(x, W, target,
                                                    part_m, part_s, t_logit);
    lmhead_reduce_rows<<<ROWS / 8, 256, 0, stream>>>(part_m, part_s, t_logit,
                                                     target, wlogp, wmask);
    lmhead_seq_avg<<<B_SEQ, 256, 0, stream>>>(wlogp, wmask, seqavg);
    lmhead_final<<<1, 32, 0, stream>>>(seqavg, (float*)d_out);
}
